// BNAFLayer_14439680049601
// MI455X (gfx1250) — compile-verified, matrix operand order checked by reading, not run on hardware
//
#include <hip/hip_runtime.h>

// ---- types matching gfx1250 WMMA builtin signatures ----
typedef __attribute__((ext_vector_type(16))) __bf16 v16bf;
typedef __attribute__((ext_vector_type(8)))  __bf16 v8bf;
typedef __attribute__((ext_vector_type(8)))  float  v8f;

#define BW_TOTAL 16384   // B*W = 256*64 rows
#define E_DIM    128     // W_IN
#define H_DIM    256     // 2*W_IN hidden
#define IDIM     64
#define ODIM     64
#define MROWS    32      // rows per block (2 x 16-row M-halves)

#define LOG2E 1.4426950408889634f
#define LN2   0.6931471805599453f

union V16U { v16bf v; v8bf h[2]; };

// fast exp(x) = 2^(x*log2e) via native v_exp_f32 (args bounded, no range fixup)
__device__ __forceinline__ float fast_exp(float x) {
  return __builtin_amdgcn_exp2f(x * LOG2E);
}
// fast ln(x) via native v_log_f32 (log2) * ln2
__device__ __forceinline__ float fast_log(float x) {
  return __builtin_amdgcn_logf(x) * LN2;
}
// branch-free tanh(x) = 1 - 2/(exp2(2*log2e*x)+1); fine at bf16 precision
__device__ __forceinline__ float fast_tanh(float x) {
  float t = __builtin_amdgcn_exp2f(x * (2.0f * LOG2E));
  return 1.0f - 2.0f * __builtin_amdgcn_rcpf(t + 1.0f);
}

// Load a 16-element bf16 fragment (A or B operand, 16-bit layout per ISA 7.12.2):
// element j -> K = (j<8 ? j : j+8) + 8*lanehalf, i.e. two runs of 8 consecutive
// bf16 (16B each) -> two *_load_b128.
__device__ __forceinline__ v16bf load_frag(const __bf16* row, int k0, int lh) {
  V16U u;
  u.h[0] = *(const v8bf*)(row + k0 + lh * 8);
  u.h[1] = *(const v8bf*)(row + k0 + 16 + lh * 8);
  return u.v;
}

// Same fragment but sourced from f32 memory (embeddings), converted to bf16.
__device__ __forceinline__ v16bf load_frag_f32(const float* row, int k0, int lh) {
  const float* p1 = row + k0 + lh * 8;
  const float* p2 = row + k0 + 16 + lh * 8;
  v16bf a;
#pragma unroll
  for (int j = 0; j < 8; ++j) {
    a[j]     = (__bf16)p1[j];
    a[8 + j] = (__bf16)p2[j];
  }
  return a;
}

__device__ __forceinline__ v8f wmma_bf16(v16bf a, v16bf b, v8f c) {
  return __builtin_amdgcn_wmma_f32_16x16x32_bf16(false, a, false, b, (short)0, c,
                                                 false, false);
}

// ---------- pre-pass: f32 -> bf16 weight conversion into workspace ----------
__global__ void cvt_f32_to_bf16(const float* __restrict__ src,
                                __bf16* __restrict__ dst, int n) {
  int i = blockIdx.x * 256 + threadIdx.x;
  if (i < n) dst[i] = (__bf16)src[i];
}

// ---------- fused main kernel ----------
// grid: 512 blocks (32 rows each), block: 256 threads (8 waves).
// wave w: M-half mh = w>>2 (16 rows), output o-quarter o0 = (w&3)*16.
// Waves w and w+4 fetch identical w_w2 B-tiles -> WGP$ sharing halves L2 traffic.
// launch_bounds(256, 4): >=4 waves/EU -> <=256 VGPRs/wave (no vgpr-msb path).
__global__ __launch_bounds__(256, 4) void bnaf_fused(
    const float* __restrict__ input, const float* __restrict__ w_emb,
    const float* __restrict__ logj, const float* __restrict__ w_b1,
    const float* __restrict__ w_b2, const float* __restrict__ b_b1,
    const float* __restrict__ b_b2, const __bf16* __restrict__ w_w1_bf,
    const __bf16* __restrict__ b_w1_bf, const __bf16* __restrict__ w_w2_bf,
    const __bf16* __restrict__ b_w2_bf, float* __restrict__ out) {
  __shared__ __bf16 H_s[MROWS * H_DIM];    // tanh(E@w_w1^T) tile, bf16 (16 KB)
  __shared__ __bf16 Hb_s[MROWS * H_DIM];   // tanh(E@b_w1^T) tile, bf16 (16 KB)
  __shared__ float in_s[MROWS * IDIM];     // input tile (8 KB)
  __shared__ float ej_s[MROWS * IDIM];     // exp(logj) tile (8 KB)

  const int tid  = threadIdx.x;
  const int wave = tid >> 5;
  const int lane = tid & 31;
  const int ll   = lane & 15;      // row (A) / col (B,C,D) selector
  const int lh   = lane >> 4;      // half-wave
  const int mh   = wave >> 2;      // which 16-row M-half
  const int q    = wave & 3;       // o-quarter index
  const int m0   = blockIdx.x * MROWS;
  const int mrow = 16 * mh;        // M-half base row within tile

  // ---- stage input / exp(logj) tiles (each tile = 2048 contiguous f32) ----
  for (int e = tid; e < MROWS * IDIM; e += 256) {
    in_s[e] = input[(size_t)m0 * IDIM + e];
    ej_s[e] = fast_exp(logj[(size_t)m0 * IDIM + e]);
  }

  // ---- embedding A-fragments for this wave's M-half, K=128 -> 4 K-steps ----
  const float* rowE = w_emb + (size_t)(m0 + mrow + ll) * E_DIM;
  v16bf ae[4];
#pragma unroll
  for (int ks = 0; ks < 4; ++ks) ae[ks] = load_frag_f32(rowE, ks * 32, lh);

  // ---- Phase 1: H = tanh(E@w_w1^T + w_b1), Hb likewise.
  // 64 jobs = 2 M-halves x 32 N-tiles; wave w does jobs t=(w&3)*8+j for half mh.
  // unroll 4: keep <=16 WMMAs of operands in flight (VGPR pressure control).
#pragma unroll 4
  for (int j = 0; j < 8; ++j) {
    const int  t   = q * 8 + j;
    const bool isW = (t < 16);
    const int  n0  = (isW ? t : t - 16) * 16;
    const __bf16* Wb = (isW ? w_w1_bf : b_w1_bf) + (size_t)(n0 + ll) * E_DIM;
    v8f c = {};
#pragma unroll
    for (int ks = 0; ks < 4; ++ks) {
      v16bf b = load_frag(Wb, ks * 32, lh);
      c = wmma_bf16(ae[ks], b, c);
    }
    const float bs  = (isW ? w_b1 : b_b1)[n0 + ll];
    __bf16*     dst = isW ? H_s : Hb_s;
#pragma unroll
    for (int r = 0; r < 8; ++r) {
      float h = fast_tanh(c[r] + bs);
      dst[(mrow + r + 8 * lh) * H_DIM + n0 + ll] = (__bf16)h;
    }
  }
  __syncthreads();

  const int o0 = q * 16;

  // ---- Phase 2: b1 tile = Hb @ b_w2^T  (K=256 -> 8 K-steps) ----
  v8f b1acc = {};
#pragma unroll
  for (int ks = 0; ks < 8; ++ks) {
    v16bf a = load_frag(Hb_s + (mrow + ll) * H_DIM, ks * 32, lh);
    v16bf b = load_frag(b_w2_bf + (size_t)(o0 + ll) * H_DIM, ks * 32, lh);
    b1acc = wmma_bf16(a, b, b1acc);
  }

  // ---- preload H A-fragments once; reused for all 64 i-iterations ----
  v16bf aH[8];
#pragma unroll
  for (int ks = 0; ks < 8; ++ks)
    aH[ks] = load_frag(H_s + (mrow + ll) * H_DIM, ks * 32, lh);

  float out_acc[8], ssum[8];
#pragma unroll
  for (int r = 0; r < 8; ++r) { out_acc[r] = 0.f; ssum[r] = 0.f; }

  // ---- Phase 3: stream i = 0..63; w1 tile never leaves registers ----
  for (int i = 0; i < IDIM; ++i) {
    const __bf16* Wr = w_w2_bf + (size_t)(i * ODIM + o0 + ll) * H_DIM;
    v8f c = {};
#pragma unroll
    for (int ks = 0; ks < 8; ++ks) {
      v16bf b = load_frag(Wr, ks * 32, lh);
      c = wmma_bf16(aH[ks], b, c);
    }
    if (i + 1 < IDIM)
      __builtin_prefetch(w_w2_bf + (size_t)((i + 1) * ODIM + o0 + ll) * H_DIM, 0, 0);
    const float bias = w_b2[i * ODIM + o0 + ll];
#pragma unroll
    for (int r = 0; r < 8; ++r) {
      const int   M   = mrow + r + 8 * lh;
      const float ev  = fast_exp(c[r] + bias);             // one v_exp_f32/elem
      out_acc[r] = fmaf(in_s[M * IDIM + i], ev, out_acc[r]);
      ssum[r]    = fmaf(ej_s[M * IDIM + i], ev, ssum[r]);  // exp(w1+lj)=ev*exp(lj)
    }
  }

  // ---- epilogue stores: d_out = [output | logj_out], each 16384x64 f32 ----
  const float bb2 = b_b2[o0 + ll];
  float* outp = out;
  float* ljp  = out + (size_t)BW_TOTAL * ODIM;
#pragma unroll
  for (int r = 0; r < 8; ++r) {
    const int row = m0 + mrow + r + 8 * lh;
    outp[(size_t)row * ODIM + o0 + ll] = out_acc[r] + b1acc[r] + bb2;
    ljp[(size_t)row * ODIM + o0 + ll]  = fast_log(ssum[r]);
  }
}

// ---------- workspace layout (bf16 elements) ----------
#define OFF_WW1 0                               // 256*128
#define OFF_BW1 (256 * 128)                     // 256*128
#define OFF_WW2 (2 * 256 * 128)                 // 4096*256
#define OFF_BW2 (2 * 256 * 128 + 4096 * 256)    // 64*256

extern "C" void kernel_launch(void* const* d_in, const int* in_sizes, int n_in,
                              void* d_out, int out_size, void* d_ws,
                              size_t ws_size, hipStream_t stream) {
  const float* input = (const float*)d_in[0];
  const float* w_emb = (const float*)d_in[1];
  const float* logj  = (const float*)d_in[2];
  const float* w_w1  = (const float*)d_in[3];
  const float* w_b1  = (const float*)d_in[4];
  const float* w_w2  = (const float*)d_in[5];
  const float* w_b2  = (const float*)d_in[6];
  const float* b_w1  = (const float*)d_in[7];
  const float* b_b1  = (const float*)d_in[8];
  const float* b_w2  = (const float*)d_in[9];
  const float* b_b2  = (const float*)d_in[10];
  float*  out = (float*)d_out;
  __bf16* ws  = (__bf16*)d_ws;

  cvt_f32_to_bf16<<<(256 * 128 + 255) / 256, 256, 0, stream>>>(w_w1, ws + OFF_WW1,
                                                               256 * 128);
  cvt_f32_to_bf16<<<(256 * 128 + 255) / 256, 256, 0, stream>>>(b_w1, ws + OFF_BW1,
                                                               256 * 128);
  cvt_f32_to_bf16<<<(4096 * 256 + 255) / 256, 256, 0, stream>>>(w_w2, ws + OFF_WW2,
                                                                4096 * 256);
  cvt_f32_to_bf16<<<(64 * 256 + 255) / 256, 256, 0, stream>>>(b_w2, ws + OFF_BW2,
                                                              64 * 256);

  bnaf_fused<<<BW_TOTAL / MROWS, 256, 0, stream>>>(
      input, w_emb, logj, w_b1, w_b2, b_b1, b_b2, ws + OFF_WW1, ws + OFF_BW1,
      ws + OFF_WW2, ws + OFF_BW2, out);
}
